// RetNetModule_46325517255108
// MI455X (gfx1250) — matrix operand order repaired and checked by reading, not compiled, not run on hardware
//
#include <hip/hip_runtime.h>
#include <math.h>

// RetNet forward for MI455X (gfx1250): all GEMMs via v_wmma_f32_16x16x32_bf16.
// Shapes hardwired to the reference: V=32000 D=1024 L=6 H=2 B=2 S=2048.

#define V_SZ  32000
#define D_SZ  1024
#define L_SZ  6
#define H_SZ  2
#define B_SZ  2
#define S_SZ  2048
#define DK_SZ 512
#define DV_SZ 1024
#define F_SZ  4096

typedef __attribute__((ext_vector_type(16))) __bf16        v16bf_t;
typedef __attribute__((ext_vector_type(8)))  float         v8f_t;
typedef __attribute__((ext_vector_type(8)))  unsigned int  v8u_t;

__device__ __forceinline__ unsigned short f2bf(float f) {
  unsigned int u = __float_as_uint(f);
  u += 0x7FFFu + ((u >> 16) & 1u);            // round-to-nearest-even
  return (unsigned short)(u >> 16);
}
__device__ __forceinline__ float bf2f(unsigned short h) {
  return __uint_as_float(((unsigned int)h) << 16);
}

#define TSTR 40  // LDS tile row stride (bf16 elems): 80B rows -> 16B aligned, bank-spread

// A fragment (16x32 bf16): lane m=lane&15; lanes 0-15 K=0..15, 16-31 K=16..31
__device__ __forceinline__ v16bf_t frag_a_lds(const unsigned short* tile, int row, int lane) {
  const unsigned int* p = (const unsigned int*)(tile + row * TSTR);
  int hi = (lane & 16) ? 8 : 0;
  v8u_t u;
  u[0] = p[0 + hi]; u[1] = p[1 + hi]; u[2] = p[2 + hi]; u[3] = p[3 + hi];
  u[4] = p[4 + hi]; u[5] = p[5 + hi]; u[6] = p[6 + hi]; u[7] = p[7 + hi];
  return __builtin_bit_cast(v16bf_t, u);
}
// B fragment (32x16 bf16, sourced from Bt row = output column): lanes 0-15 K=0..15, 16-31 K=16..31
__device__ __forceinline__ v16bf_t frag_b_lds(const unsigned short* tile, int row, int lane) {
  const unsigned int* p = (const unsigned int*)(tile + row * TSTR);
  int hi = (lane & 16) ? 8 : 0;
  v8u_t u;
#pragma unroll
  for (int v = 0; v < 8; ++v) u[v] = p[v + hi];
  return __builtin_bit_cast(v16bf_t, u);
}

__device__ __forceinline__ v8f_t wmma_bf16(v16bf_t a, v16bf_t b, v8f_t c) {
  return __builtin_amdgcn_wmma_f32_16x16x32_bf16(false, a, false, b, (short)0, c, false, false);
}

// ---------------------------------------------------------------------------
// Core 128x128 block-tile MAC loop. 8 waves, each wave computes a 32x64 tile
// (8 accumulators; per K-step: 2 A-frags reused over 4 B-frags -> 8 WMMAs).
// ---------------------------------------------------------------------------
__device__ __forceinline__ void mm_tile128(
    const unsigned short* __restrict__ Ab, int lda,
    const unsigned short* __restrict__ Bb, int ldb,
    int bm, int bn, int K,
    unsigned short* As, unsigned short* Bs,
    v8f_t c[8], int tid, int lane, int wm, int wn) {
  const int r = tid >> 2, gp = tid & 3;
  for (int k0 = 0; k0 < K; k0 += 32) {
    uint4 a0v = *((const uint4*)(Ab + (unsigned long long)(bm + r) * lda + k0) + gp);
    uint4 a1v = *((const uint4*)(Ab + (unsigned long long)(bm + 64 + r) * lda + k0) + gp);
    uint4 b0v = *((const uint4*)(Bb + (unsigned long long)(bn + r) * ldb + k0) + gp);
    uint4 b1v = *((const uint4*)(Bb + (unsigned long long)(bn + 64 + r) * ldb + k0) + gp);
    if (k0 + 64 < K) {  // warm GL2 for the tile after next (global_prefetch_b8)
      __builtin_prefetch((const void*)(Ab + (unsigned long long)(bm + r) * lda + k0 + 64), 0, 1);
      __builtin_prefetch((const void*)(Bb + (unsigned long long)(bn + r) * ldb + k0 + 64), 0, 1);
    }
    __syncthreads();
    *(uint4*)(As + r * TSTR + gp * 8)        = a0v;
    *(uint4*)(As + (64 + r) * TSTR + gp * 8) = a1v;
    *(uint4*)(Bs + r * TSTR + gp * 8)        = b0v;
    *(uint4*)(Bs + (64 + r) * TSTR + gp * 8) = b1v;
    __syncthreads();
    v16bf_t a0 = frag_a_lds(As, wm * 32 + (lane & 15), lane);
    v16bf_t a1 = frag_a_lds(As, wm * 32 + 16 + (lane & 15), lane);
#pragma unroll
    for (int j = 0; j < 4; ++j) {
      v16bf_t bj = frag_b_lds(Bs, wn * 64 + j * 16 + (lane & 15), lane);
      c[2 * j]     = wmma_bf16(a0, bj, c[2 * j]);
      c[2 * j + 1] = wmma_bf16(a1, bj, c[2 * j + 1]);
    }
  }
}

// ---------------------------------------------------------------------------
// Generic batched GEMM: C[M,N] = A[M,K](bf16) * Bt[N,K](bf16)^T (+bias,+gelu,+resid)
// Requirements: M%128==0, N%128==0, K%32==0, lda/ldb multiples of 8.
// ---------------------------------------------------------------------------
__global__ __launch_bounds__(256) void k_gemm(
    const unsigned short* __restrict__ A, unsigned long long sA, int lda,
    const unsigned short* __restrict__ Bt, unsigned long long sB, int ldb,
    float* __restrict__ C, unsigned short* __restrict__ Cbf,
    unsigned long long sC, int ldc, int K,
    const float* __restrict__ bias, int act,
    const float* __restrict__ resid) {
  __shared__ unsigned short As[128 * TSTR];
  __shared__ unsigned short Bs[128 * TSTR];
  const int tid = threadIdx.x, lane = tid & 31, wv = tid >> 5;
  const int wm = wv & 3, wn = wv >> 2;
  const int bm = blockIdx.y * 128, bn = blockIdx.x * 128;
  const unsigned long long z = blockIdx.z;
  v8f_t c[8];
#pragma unroll
  for (int i = 0; i < 8; ++i) c[i] = (v8f_t){0.f, 0.f, 0.f, 0.f, 0.f, 0.f, 0.f, 0.f};
  mm_tile128(A + z * sA, lda, Bt + z * sB, ldb, bm, bn, K, As, Bs, c, tid, lane, wm, wn);
#pragma unroll
  for (int j = 0; j < 4; ++j) {
    const int n = bn + wn * 64 + j * 16 + (lane & 15);
    const float bv = bias ? bias[n] : 0.f;
#pragma unroll
    for (int half = 0; half < 2; ++half) {
      v8f_t acc = c[2 * j + half];
      const int m0 = bm + wm * 32 + half * 16 + ((lane & 16) ? 8 : 0);
#pragma unroll
      for (int i = 0; i < 8; ++i) {
        int m = m0 + i;
        float v = acc[i] + bv;
        if (act == 1)  // exact GELU
          v = 0.5f * v * (1.f + erff(v * 0.70710678118654752f));
        unsigned long long idx = z * sC + (unsigned long long)m * ldc + n;
        if (resid) v += resid[idx];
        if (C)   C[idx] = v;
        if (Cbf) Cbf[idx] = f2bf(v);
      }
    }
  }
}

// ---------------------------------------------------------------------------
// qk kernel: W[z,s,t] = (qr[z,s,:] . kr[z,t,:]) * mask(h,s,t), stored bf16.
// mask = exp(decay*(s-t)) / sqrt(sum_t mask), causal; geometric row-sum closed form.
// ---------------------------------------------------------------------------
__global__ __launch_bounds__(256) void k_qk(
    const unsigned short* __restrict__ Q, const unsigned short* __restrict__ Kr,
    unsigned short* __restrict__ Wb) {
  __shared__ unsigned short As[128 * TSTR];
  __shared__ unsigned short Bs[128 * TSTR];
  const int tid = threadIdx.x, lane = tid & 31, wv = tid >> 5;
  const int wm = wv & 3, wn = wv >> 2;
  const int bm = blockIdx.y * 128, bn = blockIdx.x * 128;
  const int z = blockIdx.z;
  unsigned short* Wz = Wb + (unsigned long long)z * S_SZ * S_SZ;
  if (bn >= bm + 128) {  // strictly above the causal diagonal -> zero-fill tile
    for (int i = tid; i < 128 * 128; i += 256) {
      int rr = i >> 7, cc = i & 127;
      Wz[(unsigned long long)(bm + rr) * S_SZ + bn + cc] = 0;
    }
    return;
  }
  v8f_t c[8];
#pragma unroll
  for (int i = 0; i < 8; ++i) c[i] = (v8f_t){0.f, 0.f, 0.f, 0.f, 0.f, 0.f, 0.f, 0.f};
  mm_tile128(Q + (unsigned long long)z * S_SZ * DK_SZ, DK_SZ,
             Kr + (unsigned long long)z * S_SZ * DK_SZ, DK_SZ,
             bm, bn, DK_SZ, As, Bs, c, tid, lane, wm, wn);
  const int h = z & (H_SZ - 1);
  const float dec = logf(1.f - exp2f(-5.f - (float)h));  // < 0
  const float emr = expf(dec);
#pragma unroll
  for (int j = 0; j < 4; ++j) {
    const int t = bn + wn * 64 + j * 16 + (lane & 15);
#pragma unroll
    for (int half = 0; half < 2; ++half) {
      v8f_t acc = c[2 * j + half];
      const int s0 = bm + wm * 32 + half * 16 + ((lane & 16) ? 8 : 0);
#pragma unroll
      for (int i = 0; i < 8; ++i) {
        int s = s0 + i;
        float msum = (1.f - expf(dec * (float)(s + 1))) / (1.f - emr);
        float invs = rsqrtf(msum);
        float w = (t <= s) ? acc[i] * expf(dec * (float)(s - t)) * invs : 0.f;
        Wz[(unsigned long long)s * S_SZ + t] = f2bf(w);
      }
    }
  }
}

// ---------------------------------------------------------------------------
// LayerNorm (row = D elems) -> bf16
// ---------------------------------------------------------------------------
__global__ __launch_bounds__(256) void k_ln(const float* __restrict__ x,
                                            const float* __restrict__ w,
                                            const float* __restrict__ b,
                                            unsigned short* __restrict__ out, int Dn) {
  __shared__ float rs[256], rq[256];
  const int row = blockIdx.x, tid = threadIdx.x;
  const float* xr = x + (unsigned long long)row * Dn;
  float s = 0.f, q = 0.f;
  for (int i = tid; i < Dn; i += 256) { float v = xr[i]; s += v; q += v * v; }
  rs[tid] = s; rq[tid] = q; __syncthreads();
  for (int off = 128; off > 0; off >>= 1) {
    if (tid < off) { rs[tid] += rs[tid + off]; rq[tid] += rq[tid + off]; }
    __syncthreads();
  }
  float mu  = rs[0] / (float)Dn;
  float var = rq[0] / (float)Dn - mu * mu;
  float inv = rsqrtf(var + 1e-5f);
  unsigned short* orow = out + (unsigned long long)row * Dn;
  for (int i = tid; i < Dn; i += 256)
    orow[i] = f2bf((xr[i] - mu) * inv * w[i] + b[i]);
}

// ---------------------------------------------------------------------------
// Embedding gather
// ---------------------------------------------------------------------------
__global__ void k_embed(const int* __restrict__ ids, const float* __restrict__ emb,
                        float* __restrict__ x, int total) {
  int idx = blockIdx.x * 256 + threadIdx.x;
  if (idx >= total) return;
  int d = idx & (D_SZ - 1);
  int n = idx >> 10;
  x[idx] = emb[(unsigned long long)ids[n] * D_SZ + d];
}

// ---------------------------------------------------------------------------
// Rotary: (B,S,H,DK) f32 -> (B,H,S,DK) bf16, with scale folded in (k scaling)
// ---------------------------------------------------------------------------
__global__ void k_rope(const float* __restrict__ in, unsigned short* __restrict__ out,
                       float scale, int total_pairs) {
  int idx = blockIdx.x * 256 + threadIdx.x;
  if (idx >= total_pairs) return;
  int p = idx;
  int j = p % (DK_SZ / 2); p /= (DK_SZ / 2);
  int h = p % H_SZ;        p /= H_SZ;
  int s = p % S_SZ;
  int b = p / S_SZ;
  unsigned long long src = (((unsigned long long)(b * S_SZ + s) * H_SZ + h) * DK_SZ) + 2 * j;
  float x1 = in[src], x2 = in[src + 1];
  float ang = powf(10000.f, -(float)j / (float)(DK_SZ / 2 - 1));
  float th  = (float)s * ang;
  float sn = sinf(th), cs = cosf(th);
  float o1 = (x1 * cs - x2 * sn) * scale;
  float o2 = (x2 * cs + x1 * sn) * scale;
  unsigned long long dst = ((((unsigned long long)b * H_SZ + h) * S_SZ + s) * DK_SZ) + 2 * j;
  out[dst] = f2bf(o1);
  out[dst + 1] = f2bf(o2);
}

// v: (B,S,H,DV) f32 -> (B,H,DV,S) bf16  (so OV GEMM sees Bt=[N=DV,K=S])
__global__ void k_vT(const float* __restrict__ v, unsigned short* __restrict__ vT, int total) {
  int idx = blockIdx.x * 256 + threadIdx.x;
  if (idx >= total) return;
  int p = idx;
  int s = p % S_SZ;  p /= S_SZ;
  int d = p % DV_SZ; p /= DV_SZ;
  int h = p % H_SZ;
  int b = p / H_SZ;
  vT[idx] = f2bf(v[(unsigned long long)(b * S_SZ + s) * (H_SZ * DV_SZ) + h * DV_SZ + d]);
}

// ---------------------------------------------------------------------------
// Attention post: denom (recomputed from W row -> deterministic), RMS norm, SiLU gate.
// o_un: (B,H,S,DV) f32, Wb: (B,H,S,S) bf16, g: (B,S,2D) f32, og out: (B,S,2D) bf16
// ---------------------------------------------------------------------------
__global__ __launch_bounds__(256) void k_post(const float* __restrict__ o_un,
                                              const unsigned short* __restrict__ Wb,
                                              const float* __restrict__ g,
                                              unsigned short* __restrict__ og) {
  __shared__ float red[256];
  const int zrow = blockIdx.x;  // (b*H + h)*S + s
  const int tid = threadIdx.x;
  const int s  = zrow % S_SZ;
  const int zz = zrow / S_SZ;
  const int h  = zz % H_SZ;
  const int b  = zz / H_SZ;
  const unsigned short* wr = Wb + (unsigned long long)zrow * S_SZ;
  float asum = 0.f;
  for (int i = tid; i < S_SZ; i += 256) asum += fabsf(bf2f(wr[i]));
  red[tid] = asum; __syncthreads();
  for (int off = 128; off > 0; off >>= 1) {
    if (tid < off) red[tid] += red[tid + off];
    __syncthreads();
  }
  float den = fminf(fmaxf(red[0], 1.f), 50000.f);
  float inv = 1.f / den;
  __syncthreads();
  const float* orow = o_un + (unsigned long long)zrow * DV_SZ;
  float vals[4]; float sq = 0.f;
#pragma unroll
  for (int jj = 0; jj < 4; ++jj) {
    float vv = orow[tid + jj * 256] * inv;
    vals[jj] = vv; sq += vv * vv;
  }
  red[tid] = sq; __syncthreads();
  for (int off = 128; off > 0; off >>= 1) {
    if (tid < off) red[tid] += red[tid + off];
    __syncthreads();
  }
  float sc = rsqrtf(red[0] / (float)DV_SZ + 1e-6f);
  unsigned long long gbase = (unsigned long long)(b * S_SZ + s) * (2 * D_SZ) + h * DV_SZ;
#pragma unroll
  for (int jj = 0; jj < 4; ++jj) {
    int i = tid + jj * 256;
    float gv = g[gbase + i];
    float sig = 1.f / (1.f + expf(-gv));
    og[gbase + i] = f2bf(gv * sig * vals[jj] * sc);
  }
}

// ---------------------------------------------------------------------------
// Weight convert+transpose: W f32 (K,N) -> Wt bf16 (N,K) via 32x32 LDS tile
// ---------------------------------------------------------------------------
__global__ __launch_bounds__(256) void k_wtrans(const float* __restrict__ W,
                                                unsigned short* __restrict__ Wt,
                                                int Kd, int Nd) {
  __shared__ float t[32][33];
  int n0 = blockIdx.x * 32, k0 = blockIdx.y * 32;
  int tx = threadIdx.x & 31, ty = threadIdx.x >> 5;
  for (int yy = ty; yy < 32; yy += 8)
    t[yy][tx] = W[(unsigned long long)(k0 + yy) * Nd + n0 + tx];
  __syncthreads();
  for (int yy = ty; yy < 32; yy += 8)
    Wt[(unsigned long long)(n0 + yy) * Kd + k0 + tx] = f2bf(t[tx][yy]);
}

// ---------------------------------------------------------------------------
extern "C" void kernel_launch(void* const* d_in, const int* in_sizes, int n_in,
                              void* d_out, int out_size, void* d_ws, size_t ws_size,
                              hipStream_t stream) {
  const int*   ids   = (const int*)d_in[0];
  const float* embed = (const float*)d_in[1];
  const float* Wq    = (const float*)d_in[2];
  const float* Wk    = (const float*)d_in[3];
  const float* Wv    = (const float*)d_in[4];
  const float* Wg    = (const float*)d_in[5];
  const float* Wo    = (const float*)d_in[6];
  const float* ln1w  = (const float*)d_in[7];
  const float* ln1b  = (const float*)d_in[8];
  const float* ln2w  = (const float*)d_in[9];
  const float* ln2b  = (const float*)d_in[10];
  const float* fc1w  = (const float*)d_in[11];
  const float* fc1b  = (const float*)d_in[12];
  const float* fc2w  = (const float*)d_in[13];
  const float* fc2b  = (const float*)d_in[14];
  const float* lnfw  = (const float*)d_in[15];
  const float* lnfb  = (const float*)d_in[16];
  const float* Wout  = (const float*)d_in[17];
  float* out = (float*)d_out;
  (void)in_sizes; (void)n_in; (void)out_size; (void)ws_size;

  const int BS = B_SZ * S_SZ;  // 4096

  char* p = (char*)d_ws;
  auto carve = [&](size_t bytes) -> char* {
    char* r = p;
    p += (bytes + 255) & ~((size_t)255);
    return r;
  };
  float*          x    = (float*)carve((size_t)BS * D_SZ * 4);
  unsigned short* hbf  = (unsigned short*)carve((size_t)BS * D_SZ * 2);
  float*          qf   = (float*)carve((size_t)BS * D_SZ * 4);
  float*          kf   = (float*)carve((size_t)BS * D_SZ * 4);
  float*          vf   = (float*)carve((size_t)BS * 2 * D_SZ * 4);
  float*          gf   = (float*)carve((size_t)BS * 2 * D_SZ * 4);
  unsigned short* qr   = (unsigned short*)carve((size_t)BS * D_SZ * 2);
  unsigned short* kr   = (unsigned short*)carve((size_t)BS * D_SZ * 2);
  unsigned short* vT   = (unsigned short*)carve((size_t)BS * 2 * D_SZ * 2);
  unsigned short* Wb   = (unsigned short*)carve((size_t)B_SZ * H_SZ * S_SZ * S_SZ * 2);
  float*          o_un = (float*)carve((size_t)B_SZ * H_SZ * S_SZ * DV_SZ * 4);
  unsigned short* og   = (unsigned short*)carve((size_t)BS * 2 * D_SZ * 2);
  unsigned short* fbf  = (unsigned short*)carve((size_t)BS * F_SZ * 2);
  unsigned short* wt   = (unsigned short*)carve((size_t)D_SZ * V_SZ * 2);

  auto gemm = [&](const unsigned short* A, size_t sA, int lda,
                  const unsigned short* Bt, size_t sB, int ldb,
                  float* C, unsigned short* Cbf, size_t sC, int ldc,
                  int M, int N, int Kd, int batch,
                  const float* bias, int act, const float* resid) {
    dim3 g(N / 128, M / 128, batch);
    k_gemm<<<g, 256, 0, stream>>>(A, sA, lda, Bt, sB, ldb, C, Cbf, sC, ldc, Kd, bias, act, resid);
  };
  auto wtrans = [&](const float* W, int Kd, int Nd) {
    dim3 g(Nd / 32, Kd / 32);
    k_wtrans<<<g, 256, 0, stream>>>(W, wt, Kd, Nd);
  };

  k_embed<<<(BS * D_SZ) / 256, 256, 0, stream>>>(ids, embed, x, BS * D_SZ);

  const float scaling = 0.04419417382415922f;  // DK^-0.5
  for (int l = 0; l < L_SZ; ++l) {
    k_ln<<<BS, 256, 0, stream>>>(x, ln1w + l * D_SZ, ln1b + l * D_SZ, hbf, D_SZ);

    wtrans(Wq + (size_t)l * D_SZ * D_SZ, D_SZ, D_SZ);
    gemm(hbf, 0, D_SZ, wt, 0, D_SZ, qf, nullptr, 0, D_SZ, BS, D_SZ, D_SZ, 1, nullptr, 0, nullptr);
    wtrans(Wk + (size_t)l * D_SZ * D_SZ, D_SZ, D_SZ);
    gemm(hbf, 0, D_SZ, wt, 0, D_SZ, kf, nullptr, 0, D_SZ, BS, D_SZ, D_SZ, 1, nullptr, 0, nullptr);
    wtrans(Wv + (size_t)l * D_SZ * 2 * D_SZ, D_SZ, 2 * D_SZ);
    gemm(hbf, 0, D_SZ, wt, 0, D_SZ, vf, nullptr, 0, 2 * D_SZ, BS, 2 * D_SZ, D_SZ, 1, nullptr, 0, nullptr);
    wtrans(Wg + (size_t)l * D_SZ * 2 * D_SZ, D_SZ, 2 * D_SZ);
    gemm(hbf, 0, D_SZ, wt, 0, D_SZ, gf, nullptr, 0, 2 * D_SZ, BS, 2 * D_SZ, D_SZ, 1, nullptr, 0, nullptr);

    const int pairs = BS * D_SZ / 2;
    k_rope<<<pairs / 256, 256, 0, stream>>>(qf, qr, 1.f, pairs);
    k_rope<<<pairs / 256, 256, 0, stream>>>(kf, kr, scaling, pairs);
    const int vtot = BS * 2 * D_SZ;
    k_vT<<<vtot / 256, 256, 0, stream>>>(vf, vT, vtot);

    dim3 gq(S_SZ / 128, S_SZ / 128, B_SZ * H_SZ);
    k_qk<<<gq, 256, 0, stream>>>(qr, kr, Wb);

    gemm(Wb, (size_t)S_SZ * S_SZ, S_SZ, vT, (size_t)DV_SZ * S_SZ, S_SZ,
         o_un, nullptr, (size_t)S_SZ * DV_SZ, DV_SZ,
         S_SZ, DV_SZ, S_SZ, B_SZ * H_SZ, nullptr, 0, nullptr);

    k_post<<<B_SZ * H_SZ * S_SZ, 256, 0, stream>>>(o_un, Wb, gf, og);

    wtrans(Wo + (size_t)l * 2 * D_SZ * D_SZ, 2 * D_SZ, D_SZ);
    gemm(og, 0, 2 * D_SZ, wt, 0, 2 * D_SZ, x, nullptr, 0, D_SZ,
         BS, D_SZ, 2 * D_SZ, 1, nullptr, 0, x);

    k_ln<<<BS, 256, 0, stream>>>(x, ln2w + l * D_SZ, ln2b + l * D_SZ, hbf, D_SZ);
    wtrans(fc1w + (size_t)l * D_SZ * F_SZ, D_SZ, F_SZ);
    gemm(hbf, 0, D_SZ, wt, 0, D_SZ, nullptr, fbf, 0, F_SZ,
         BS, F_SZ, D_SZ, 1, fc1b + l * F_SZ, 1, nullptr);
    wtrans(fc2w + (size_t)l * F_SZ * D_SZ, F_SZ, D_SZ);
    gemm(fbf, 0, F_SZ, wt, 0, F_SZ, x, nullptr, 0, D_SZ,
         BS, D_SZ, F_SZ, 1, fc2b + l * D_SZ, 0, x);
  }

  k_ln<<<BS, 256, 0, stream>>>(x, lnfw, lnfb, hbf, D_SZ);
  wtrans(Wout, D_SZ, V_SZ);
  gemm(hbf, 0, D_SZ, wt, 0, D_SZ, out, nullptr, 0, V_SZ,
       BS, V_SZ, D_SZ, 1, nullptr, 0, nullptr);
}
